// MemAttention_37580963840516
// MI455X (gfx1250) — compile-verified, matrix operand order checked by reading, not run on hardware
//
#include <hip/hip_runtime.h>
#include <cstdint>

// ---------------------------------------------------------------------------
// MI455X (gfx1250) memory-attention: bf16 WMMA GEMMs + fused flash attention.
// Wave32, 16x16x32 bf16 WMMA, f32 accumulation, async global->LDS staging.
// ---------------------------------------------------------------------------

typedef __attribute__((ext_vector_type(16))) __bf16 bf16x16;
typedef __attribute__((ext_vector_type(8)))  float  f32x8;

union FragU { unsigned int u[8]; bf16x16 v; };

__device__ __forceinline__ unsigned short f2bf(float f) {
    union { float f; unsigned u; } x; x.f = f;
    unsigned r = x.u + 0x7FFFu + ((x.u >> 16) & 1u);   // round-to-nearest-even
    return (unsigned short)(r >> 16);
}

__device__ __forceinline__ f32x8 wmma_bf16(bf16x16 a, bf16x16 b, f32x8 c) {
    return __builtin_amdgcn_wmma_f32_16x16x32_bf16(false, a, false, b, (short)0, c, false, false);
}

// CDNA5 async copy: 16B global -> LDS via the async engine (no VGPR roundtrip).
// LDS offset = low 32 bits of the generic pointer (flat LDS aperture encoding).
__device__ __forceinline__ void async_copy_b128(void* lds_ptr, const void* gptr) {
    unsigned loff = (unsigned)(uintptr_t)lds_ptr;
    asm volatile("global_load_async_to_lds_b128 %0, %1, off"
                 :: "v"(loff), "v"(gptr)
                 : "memory");
}

template <int N>
__device__ __forceinline__ void wait_async() {
    asm volatile("s_wait_asynccnt %0" :: "i"(N) : "memory");
}

// A-style 16x32 fragment from row-major bf16 (ushort) data.
// lane&15 = row; lane>>4 selects K-halves; each VGPR packs 2 consecutive K.
__device__ __forceinline__ bf16x16 frag_row(const unsigned short* base, int stride, int lane) {
    int m = lane & 15, hf = lane >> 4;
    FragU f;
    const unsigned short* row = base + (size_t)m * stride;
#pragma unroll
    for (int v = 0; v < 8; ++v) {
        int k = ((v & 4) << 2) + hf * 8 + ((v & 3) << 1);
        f.u[v] = *(const unsigned int*)(row + k);
    }
    return f.v;
}

// B-style 32x16 fragment where data is stored K-major (row-major V tile):
// lane&15 = output column; two K values gathered with stride.
__device__ __forceinline__ bf16x16 frag_col(const unsigned short* base, int stride, int lane) {
    int n = lane & 15, hf = lane >> 4;
    FragU f;
#pragma unroll
    for (int v = 0; v < 8; ++v) {
        int k = ((v & 4) << 2) + hf * 8 + ((v & 3) << 1);
        f.u[v] = (unsigned)base[(size_t)k * stride + n] |
                 ((unsigned)base[(size_t)(k + 1) * stride + n] << 16);
    }
    return f.v;
}

// ---------------------------------------------------------------------------
// Generic GEMM: C[M,N] = A[M,K] * W[N,K]^T (+bias)(+residual)(relu)
// A, W bf16 row-major; output fp32 or bf16. Block = 128 threads, 64x64 tile,
// double-buffered async LDS staging.
// ---------------------------------------------------------------------------
__global__ __launch_bounds__(128) void k_gemm(
    const unsigned short* __restrict__ A, const unsigned short* __restrict__ W,
    const float* __restrict__ bias, const float* residual,
    float* outF, unsigned short* outB, int M, int N, int K, int relu)
{
    __shared__ __align__(16) unsigned short sA[2][64 * 32];
    __shared__ __align__(16) unsigned short sB[2][64 * 32];
    const int tid = threadIdx.x, lane = tid & 31, wave = tid >> 5;
    const int mbase = blockIdx.y * 64, nbase = blockIdx.x * 64;
    const f32x8 fz = {0.f,0.f,0.f,0.f,0.f,0.f,0.f,0.f};
    f32x8 acc[4];
#pragma unroll
    for (int t = 0; t < 4; ++t) acc[t] = fz;

    // each thread stages 2x16B into sA and sB per tile (4 async ops)
    auto stage = [&](int buf, int kc) {
#pragma unroll
        for (int i = 0; i < 2; ++i) {
            int q = tid * 2 + i, r = q >> 2, c = (q & 3) * 8;
            async_copy_b128(&sA[buf][r * 32 + c], A + (size_t)(mbase + r) * K + kc + c);
            async_copy_b128(&sB[buf][r * 32 + c], W + (size_t)(nbase + r) * K + kc + c);
        }
    };

    const int nk = K >> 5;
    stage(0, 0);
    for (int ki = 0; ki < nk; ++ki) {
        const int cur = ki & 1;
        const bool hasNext = (ki + 1) < nk;
        if (hasNext) stage(cur ^ 1, (ki + 1) << 5);      // prefetch next tile
        if (hasNext) wait_async<4>();                    // retire current tile only
        else         wait_async<0>();
        __syncthreads();
        bf16x16 a = frag_row(sA[cur] + wave * 16 * 32, 32, lane);
#pragma unroll
        for (int t = 0; t < 4; ++t) {
            bf16x16 b = frag_row(sB[cur] + t * 16 * 32, 32, lane);
            acc[t] = wmma_bf16(a, b, acc[t]);
        }
        __syncthreads();   // all reads of 'cur' done before it is restaged
    }

    const int nl = lane & 15, r0 = (lane >> 4) * 8;
#pragma unroll
    for (int t = 0; t < 4; ++t) {
        int col = nbase + t * 16 + nl;
        float bv = bias ? bias[col] : 0.f;
#pragma unroll
        for (int j = 0; j < 8; ++j) {
            int row = mbase + wave * 16 + r0 + j;
            float v = acc[t][j] + bv;
            if (residual) v += residual[(size_t)row * N + col];
            if (relu) v = fmaxf(v, 0.f);
            if (outF) outF[(size_t)row * N + col] = v;
            else      outB[(size_t)row * N + col] = f2bf(v);
        }
    }
}

// ---------------------------------------------------------------------------
// Fused flash attention, single head dim 256. Br=16 queries/block, Bc=64.
// Q pre-scaled by 1/sqrt(256). One 32KB LDS buffer reused for K then V tiles;
// async staging overlaps the V copy with the online-softmax scalar work.
// ---------------------------------------------------------------------------
__global__ __launch_bounds__(128) void k_attn(
    const unsigned short* __restrict__ Q, const unsigned short* __restrict__ Kg,
    const unsigned short* __restrict__ Vg, unsigned short* __restrict__ O, int nkeys)
{
    __shared__ __align__(16) unsigned short sKV[64 * 256];   // 32 KB, K then V
    __shared__ __align__(16) float          sS[16 * 64];     // score tile
    __shared__ __align__(16) unsigned short sP[16 * 64];     // softmax tile (bf16)
    __shared__ float sm[16], sl[16], srs[16], sli[16];

    const int tid = threadIdx.x, lane = tid & 31, wave = tid >> 5;
    const int qbase = blockIdx.x * 16;
    const int hf = lane >> 4, nl = lane & 15, r0 = hf * 8;
    const f32x8 fz = {0.f,0.f,0.f,0.f,0.f,0.f,0.f,0.f};

    // Q fragments (16 rows x 256) pinned in registers for the whole kernel.
    bf16x16 qf[8];
    {
        const unsigned short* qrow = Q + (size_t)(qbase + nl) * 256;
#pragma unroll
        for (int kc = 0; kc < 8; ++kc) {
            FragU f;
#pragma unroll
            for (int v = 0; v < 8; ++v) {
                int k = ((v & 4) << 2) + hf * 8 + ((v & 3) << 1);
                f.u[v] = *(const unsigned int*)(qrow + kc * 32 + k);
            }
            qf[kc] = f.v;
        }
    }

    f32x8 oacc[4];
#pragma unroll
    for (int t = 0; t < 4; ++t) oacc[t] = fz;
    if (tid < 16) { sm[tid] = -3.0e38f; sl[tid] = 0.f; }

    const int nTiles = nkeys >> 6;
    for (int kb = 0; kb < nTiles; ++kb) {
        __syncthreads();
        // async-stage K tile (64 x 256 bf16), 16x16B per thread
#pragma unroll
        for (int i = 0; i < 16; ++i) {
            int id = i * 128 + tid, r = id >> 5, c = (id & 31) * 8;
            async_copy_b128(sKV + r * 256 + c,
                            Kg + ((size_t)(kb * 64 + r)) * 256 + c);
        }
        wait_async<0>();
        __syncthreads();

        // S tile: wave w computes 16x16 slice for keys [kb*64 + w*16, +16)
        f32x8 s = fz;
#pragma unroll
        for (int kc = 0; kc < 8; ++kc) {
            bf16x16 b = frag_row(sKV + wave * 16 * 256 + kc * 32, 256, lane);
            s = wmma_bf16(qf[kc], b, s);
        }
#pragma unroll
        for (int j = 0; j < 8; ++j)
            sS[(r0 + j) * 64 + wave * 16 + nl] = s[j];
        __syncthreads();

        // async-stage V tile into the same LDS (K fully consumed);
        // overlapped with the online-softmax stats below.
#pragma unroll
        for (int i = 0; i < 16; ++i) {
            int id = i * 128 + tid, r = id >> 5, c = (id & 31) * 8;
            async_copy_b128(sKV + r * 256 + c,
                            Vg + ((size_t)(kb * 64 + r)) * 256 + c);
        }
        if (tid < 16) {
            float mo = sm[tid], mx = mo;
            for (int c = 0; c < 64; ++c) mx = fmaxf(mx, sS[tid * 64 + c]);
            float rs = __expf(mo - mx), sum = 0.f;
            for (int c = 0; c < 64; ++c) {
                float p = __expf(sS[tid * 64 + c] - mx);
                sum += p;
                sP[tid * 64 + c] = f2bf(p);
            }
            sm[tid] = mx; sl[tid] = sl[tid] * rs + sum; srs[tid] = rs;
        }
        wait_async<0>();
        __syncthreads();

        // rescale accumulators, then O += P(16x64) * V(64x256) slice
#pragma unroll
        for (int t = 0; t < 4; ++t)
#pragma unroll
            for (int j = 0; j < 8; ++j) oacc[t][j] *= srs[r0 + j];
#pragma unroll
        for (int kc = 0; kc < 2; ++kc) {
            bf16x16 a = frag_row(sP + kc * 32, 64, lane);
#pragma unroll
            for (int t = 0; t < 4; ++t) {
                bf16x16 b = frag_col(sKV + kc * 32 * 256 + wave * 64 + t * 16, 256, lane);
                oacc[t] = wmma_bf16(a, b, oacc[t]);
            }
        }
    }

    __syncthreads();
    if (tid < 16) sli[tid] = 1.f / sl[tid];
    __syncthreads();
#pragma unroll
    for (int t = 0; t < 4; ++t)
#pragma unroll
        for (int j = 0; j < 8; ++j) {
            float v = oacc[t][j] * sli[r0 + j];
            O[(size_t)(qbase + r0 + j) * 256 + wave * 64 + t * 16 + nl] = f2bf(v);
        }
}

// ---------------------------------------------------------------------------
// LayerNorm over 256 cols; wave per row. Optional bf16 row-major out and/or
// fp32 transposed out (for the final (1,256,64,64) output).
// ---------------------------------------------------------------------------
__global__ __launch_bounds__(256) void k_ln(
    const float* __restrict__ X, const float* __restrict__ S, const float* __restrict__ B,
    unsigned short* outB, float* outT, int rows)
{
    int lane = threadIdx.x & 31;
    int row = blockIdx.x * 8 + (threadIdx.x >> 5);
    if (row >= rows) return;
    float v[8], s = 0.f;
#pragma unroll
    for (int i = 0; i < 8; ++i) { v[i] = X[(size_t)row * 256 + i * 32 + lane]; s += v[i]; }
#pragma unroll
    for (int off = 16; off > 0; off >>= 1) s += __shfl_xor(s, off, 32);
    float mean = s * (1.f / 256.f), var = 0.f;
#pragma unroll
    for (int i = 0; i < 8; ++i) { float d = v[i] - mean; var += d * d; }
#pragma unroll
    for (int off = 16; off > 0; off >>= 1) var += __shfl_xor(var, off, 32);
    float inv = rsqrtf(var * (1.f / 256.f) + 1e-5f);
#pragma unroll
    for (int i = 0; i < 8; ++i) {
        int col = i * 32 + lane;
        float o = (v[i] - mean) * inv * S[col] + B[col];
        if (outB) outB[(size_t)row * 256 + col] = f2bf(o);
        if (outT) outT[(size_t)col * 4096 + row] = o;
    }
}

// Axial RoPE tables: cos/sin (4096 x 128).
__global__ void k_tables(float* cosT, float* sinT)
{
    int i = blockIdx.x * blockDim.x + threadIdx.x;
    if (i >= 4096 * 128) return;
    int srow = i >> 7, col = i & 127, fi = col & 63;
    float freq = __powf(10000.f, -(4.f * (float)fi) / 256.f);
    float t = (col < 64) ? (float)(srow & 63) : (float)(srow >> 6);
    float a = t * freq;
    cosT[i] = cosf(a);
    sinT[i] = sinf(a);
}

// RoPE apply (rows x 256 f32 -> bf16), first ropeRows rotated (table mod 4096),
// rest passed through; 'scale' folds in 1/sqrt(d) for queries.
__global__ void k_rope(const float* __restrict__ src, unsigned short* __restrict__ dst,
                       const float* __restrict__ cosT, const float* __restrict__ sinT,
                       int rows, int ropeRows, float scale)
{
    int i = blockIdx.x * blockDim.x + threadIdx.x;
    if (i >= rows * 128) return;
    int row = i >> 7, col = i & 127;
    float e = src[(size_t)row * 256 + 2 * col];
    float o = src[(size_t)row * 256 + 2 * col + 1];
    float re = e, ro = o;
    if (row < ropeRows) {
        int tr = row & 4095;
        float c = cosT[tr * 128 + col], s = sinT[tr * 128 + col];
        re = e * c - o * s;
        ro = e * s + o * c;
    }
    dst[(size_t)row * 256 + 2 * col]     = f2bf(re * scale);
    dst[(size_t)row * 256 + 2 * col + 1] = f2bf(ro * scale);
}

// x = transpose(vision_feat) - no_mem_embed + 0.1 * pos
__global__ void k_build_x(const float* __restrict__ feat, const float* __restrict__ pos,
                          const float* __restrict__ nme, float* __restrict__ x)
{
    int i = blockIdx.x * blockDim.x + threadIdx.x;
    if (i >= 4096 * 256) return;
    int srow = i >> 8, d = i & 255;
    x[i] = feat[(size_t)d * 4096 + srow] - nme[d] + 0.1f * pos[i];
}

// Build mem (16448 x 64) and mem+mem_pos as bf16.
__global__ void k_build_mem(const float* __restrict__ m1, const float* __restrict__ m2,
                            const float* __restrict__ mp1, const float* __restrict__ mp2,
                            unsigned short* __restrict__ memB, unsigned short* __restrict__ mempB)
{
    int i = blockIdx.x * blockDim.x + threadIdx.x;
    if (i >= 16448 * 64) return;
    int r = i >> 6, c = i & 63;
    float val, p;
    if (r < 16384) {
        int m = r >> 12, srow = r & 4095;
        val = m2[(size_t)m * 262144 + (size_t)c * 4096 + srow];
        p = mp1[(size_t)r * 64 + c];
    } else {
        int rr = r - 16384, ii = rr >> 2, j = rr & 3;
        val = m1[(size_t)ii * 256 + j * 64 + c];
        p = mp2[(size_t)rr * 64 + c];
    }
    memB[i]  = f2bf(val);
    mempB[i] = f2bf(val + p);
}

__global__ void k_f2bf(const float* __restrict__ in, unsigned short* __restrict__ out, int n)
{
    int i = blockIdx.x * blockDim.x + threadIdx.x;
    if (i < n) out[i] = f2bf(in[i]);
}

// ---------------------------------------------------------------------------
extern "C" void kernel_launch(void* const* d_in, const int* in_sizes, int n_in,
                              void* d_out, int out_size, void* d_ws, size_t ws_size,
                              hipStream_t stream)
{
    (void)in_sizes; (void)n_in; (void)out_size; (void)ws_size;

    const float* feat = (const float*)d_in[0];
    const float* pos  = (const float*)d_in[1];
    const float* mem1 = (const float*)d_in[2];
    const float* mem2 = (const float*)d_in[3];
    const float* mp1  = (const float*)d_in[4];
    const float* mp2  = (const float*)d_in[5];
    const float* nme  = (const float*)d_in[6];
    const float* saWq = (const float*)d_in[7];  const float* sabq = (const float*)d_in[8];
    const float* saWk = (const float*)d_in[9];  const float* sabk = (const float*)d_in[10];
    const float* saWv = (const float*)d_in[11]; const float* sabv = (const float*)d_in[12];
    const float* saWo = (const float*)d_in[13]; const float* sabo = (const float*)d_in[14];
    const float* caWq = (const float*)d_in[15]; const float* cabq = (const float*)d_in[16];
    const float* caWk = (const float*)d_in[17]; const float* cabk = (const float*)d_in[18];
    const float* caWv = (const float*)d_in[19]; const float* cabv = (const float*)d_in[20];
    const float* caWo = (const float*)d_in[21]; const float* cabo = (const float*)d_in[22];
    const float* n1s = (const float*)d_in[23];  const float* n1b = (const float*)d_in[24];
    const float* n2s = (const float*)d_in[25];  const float* n2b = (const float*)d_in[26];
    const float* n3s = (const float*)d_in[27];  const float* n3b = (const float*)d_in[28];
    const float* fW1 = (const float*)d_in[29];  const float* fb1 = (const float*)d_in[30];
    const float* fW2 = (const float*)d_in[31];  const float* fb2 = (const float*)d_in[32];
    const float* fns = (const float*)d_in[33];  const float* fnb = (const float*)d_in[34];
    float* out = (float*)d_out;

    char* wsbase = (char*)d_ws;
    size_t off = 0;
    auto alloc = [&](size_t bytes) -> char* {
        char* p = wsbase + off;
        off += (bytes + 255) & ~(size_t)255;
        return p;
    };

    float*          x     = (float*)alloc(4096ull * 256 * 4);
    unsigned short* t2    = (unsigned short*)alloc(4096ull * 256 * 2);
    float*          tmpf  = (float*)alloc(16448ull * 256 * 4);
    unsigned short* qb    = (unsigned short*)alloc(4096ull * 256 * 2);
    unsigned short* kbf   = (unsigned short*)alloc(16448ull * 256 * 2);
    unsigned short* vbf   = (unsigned short*)alloc(16448ull * 256 * 2);
    unsigned short* ab    = (unsigned short*)alloc(4096ull * 256 * 2);
    unsigned short* hb    = (unsigned short*)alloc(4096ull * 2048 * 2);
    unsigned short* memB  = (unsigned short*)alloc(16448ull * 64 * 2);
    unsigned short* mempB = (unsigned short*)alloc(16448ull * 64 * 2);
    float*          cosT  = (float*)alloc(4096ull * 128 * 4);
    float*          sinT  = (float*)alloc(4096ull * 128 * 4);
    unsigned short* wsaq  = (unsigned short*)alloc(262144ull * 2);
    unsigned short* wsak  = (unsigned short*)alloc(262144ull * 2);
    unsigned short* wsav  = (unsigned short*)alloc(262144ull * 2);
    unsigned short* wsao  = (unsigned short*)alloc(262144ull * 2);
    unsigned short* wcaq  = (unsigned short*)alloc(262144ull * 2);
    unsigned short* wcak  = (unsigned short*)alloc(65536ull * 2);
    unsigned short* wcav  = (unsigned short*)alloc(65536ull * 2);
    unsigned short* wcao  = (unsigned short*)alloc(262144ull * 2);
    unsigned short* wff1  = (unsigned short*)alloc(2097152ull * 2);
    unsigned short* wff2  = (unsigned short*)alloc(2097152ull * 2);

    auto conv = [&](const float* src, unsigned short* dst, int n) {
        k_f2bf<<<dim3((n + 255) / 256), dim3(256), 0, stream>>>(src, dst, n);
    };
    auto gemm = [&](const unsigned short* A, const unsigned short* W, const float* bias,
                    const float* resid, float* oF, unsigned short* oB,
                    int M, int N, int K, int relu) {
        k_gemm<<<dim3(N / 64, M / 64), dim3(128), 0, stream>>>(A, W, bias, resid, oF, oB, M, N, K, relu);
    };
    auto rope = [&](const float* src, unsigned short* dst, int rows, int ropeRows, float scale) {
        int n = rows * 128;
        k_rope<<<dim3((n + 255) / 256), dim3(256), 0, stream>>>(src, dst, cosT, sinT, rows, ropeRows, scale);
    };
    auto ln = [&](const float* X, const float* S, const float* B,
                  unsigned short* oB, float* oT) {
        k_ln<<<dim3(512), dim3(256), 0, stream>>>(X, S, B, oB, oT, 4096);
    };

    // ---- one-time prep (re-done every call for determinism) ----
    k_tables<<<dim3((4096 * 128) / 256), dim3(256), 0, stream>>>(cosT, sinT);
    k_build_x<<<dim3((4096 * 256) / 256), dim3(256), 0, stream>>>(feat, pos, nme, x);
    k_build_mem<<<dim3((16448 * 64) / 256), dim3(256), 0, stream>>>(mem1, mem2, mp1, mp2, memB, mempB);
    conv(saWq, wsaq, 262144); conv(saWk, wsak, 262144);
    conv(saWv, wsav, 262144); conv(saWo, wsao, 262144);
    conv(caWq, wcaq, 262144); conv(caWk, wcak, 65536);
    conv(caWv, wcav, 65536);  conv(caWo, wcao, 262144);
    conv(fW1, wff1, 2097152); conv(fW2, wff2, 2097152);

    const float qscale = 1.f / 16.f;   // 1/sqrt(256)

    for (int l = 0; l < 4; ++l) {
        // ---------------- self attention ----------------
        ln(x, n1s + l * 256, n1b + l * 256, t2, nullptr);
        gemm(t2, wsaq + l * 65536, sabq + l * 256, nullptr, tmpf, nullptr, 4096, 256, 256, 0);
        rope(tmpf, qb, 4096, 4096, qscale);
        gemm(t2, wsak + l * 65536, sabk + l * 256, nullptr, tmpf, nullptr, 4096, 256, 256, 0);
        rope(tmpf, kbf, 4096, 4096, 1.f);
        gemm(t2, wsav + l * 65536, sabv + l * 256, nullptr, nullptr, vbf, 4096, 256, 256, 0);
        k_attn<<<dim3(256), dim3(128), 0, stream>>>(qb, kbf, vbf, ab, 4096);
        gemm(ab, wsao + l * 65536, sabo + l * 256, x, x, nullptr, 4096, 256, 256, 0);

        // ---------------- cross attention ----------------
        ln(x, n2s + l * 256, n2b + l * 256, t2, nullptr);
        gemm(t2, wcaq + l * 65536, cabq + l * 256, nullptr, tmpf, nullptr, 4096, 256, 256, 0);
        rope(tmpf, qb, 4096, 4096, qscale);
        gemm(mempB, wcak + l * 16384, cabk + l * 256, nullptr, tmpf, nullptr, 16448, 256, 64, 0);
        rope(tmpf, kbf, 16448, 16384, 1.f);
        gemm(memB, wcav + l * 16384, cabv + l * 256, nullptr, nullptr, vbf, 16448, 256, 64, 0);
        k_attn<<<dim3(256), dim3(128), 0, stream>>>(qb, kbf, vbf, ab, 16448);
        gemm(ab, wcao + l * 65536, cabo + l * 256, x, x, nullptr, 4096, 256, 256, 0);

        // ---------------- FFN ----------------
        ln(x, n3s + l * 256, n3b + l * 256, t2, nullptr);
        gemm(t2, wff1 + l * 524288, fb1 + l * 2048, nullptr, nullptr, hb, 4096, 2048, 256, 1);
        gemm(hb, wff2 + l * 524288, fb2 + l * 256, x, x, nullptr, 4096, 256, 2048, 0);
    }

    // final LN, transposed fp32 output (1, 256, 64, 64)
    ln(x, fns, fnb, nullptr, out);
}
